// DiagonalLSTM_23175643529578
// MI455X (gfx1250) — matrix operand order repaired and checked by reading, not compiled, not run on hardware
//
#include <hip/hip_runtime.h>

typedef _Float16 f16;
typedef __attribute__((ext_vector_type(16))) _Float16 v16h;
typedef __attribute__((ext_vector_type(8)))  _Float16 v8h;
typedef __attribute__((ext_vector_type(8)))  float    v8f;
typedef __attribute__((ext_vector_type(4)))  unsigned int v4u;
typedef __attribute__((ext_vector_type(8)))  int          v8i;
typedef __attribute__((ext_vector_type(4)))  int          v4i;

namespace {
constexpr int kB = 8, kHid = 128, kH = 64, kW = 64, kT = 127, kCin = 128;
// Fused GEMM K layout: [0,128) = x_t, [128,256) = h(y-1), [256,384) = h(y),
// [384,416) = bias block (activation==1 at K=384, weight column = b_is+b_ss).
constexpr int kNKT  = 13;    // K tiles of 32
constexpr int kNMT  = 32;    // M tiles of 16 (512 gate rows)
constexpr int kXStr = 136;   // xbuf LDS row stride (halfs): 272B = 17*16, bank stride 4
constexpr int kHStr = 296;   // hbuf LDS row stride (halfs): 592B = 37*16, bank stride 20
constexpr int kXbufHalfs = kH * kXStr;   // 8704
constexpr int kHbufHalfs = kH * kHStr;   // 18944
constexpr long kXtSlice = (long)kH * kCin;             // 8192 halfs per (b,t)
constexpr long kXtElems = (long)kB * kT * kXtSlice;    // ~16 MB
constexpr long kWfElems = (long)kNMT * kNKT * 32 * 16; // ~0.4 MB
}

// Repack x into diagonal-major f16: xT[b][t][y][c] = x[b][c][y][t-y] (0 outside band).
__global__ void dlstm_prep_x(const float* __restrict__ x, f16* __restrict__ xT) {
  long idx = (long)blockIdx.x * blockDim.x + threadIdx.x;
  if (idx >= kXtElems) return;
  int t = (int)(idx % kT); long r = idx / kT;
  int y = (int)(r % kH);   r /= kH;
  int c = (int)(r % kCin); int b = (int)(r / kCin);
  int w = t - y;
  float v = (w >= 0 && w < kW) ? x[(((long)b * kCin + c) * kH + y) * kW + w] : 0.f;
  xT[(((long)b * kT + t) * kH + y) * kCin + c] = (f16)v;
}

// Pre-swizzle fused weights into per-lane WMMA A-fragment layout, ordered [kt][mt]
// so one wave's 4 gate fragments sit at constant offsets from a per-kt pointer.
__global__ void dlstm_prep_w(const float* __restrict__ w_is, const float* __restrict__ b_is,
                             const float* __restrict__ w_ss, const float* __restrict__ b_ss,
                             f16* __restrict__ wfrag) {
  long idx = (long)blockIdx.x * blockDim.x + threadIdx.x;
  if (idx >= kWfElems) return;
  int j    = (int)(idx & 15);
  int lane = (int)((idx >> 4) & 31);
  int frag = (int)(idx >> 9);
  int mt = frag & 31;        // [kt][mt] order
  int kt = frag >> 5;
  int M   = mt * 16 + (lane & 15);
  int hiq = lane >> 4;
  int kl  = j + 16 * hiq;  // contiguous K order: k = 16*(lane/16) + j (matches B loads)
  int K   = kt * 32 + kl;
  float v = 0.f;
  if (K < 128)       v = w_is[M * 128 + K];
  else if (K < 256)  v = w_ss[(M * 128 + (K - 128)) * 2 + 0];  // tap0 -> h(y-1)
  else if (K < 384)  v = w_ss[(M * 128 + (K - 256)) * 2 + 1];  // tap1 -> h(y)
  else if (K == 384) v = b_is[M] + b_ss[M];                    // bias column
  wfrag[idx] = (f16)v;
}

__device__ __forceinline__ float fast_rcp(float x) { return __builtin_amdgcn_rcpf(x); }

// Persistent diagonal-scan kernel: one block per batch, 32 wave32 (1024 threads).
// wave (wi,wj): gate-rows {16*wi..+16} of all 4 gates x cols {16*wj..+16}.
__global__ __launch_bounds__(1024)
void dlstm_scan(const f16* __restrict__ xT, const f16* __restrict__ wfrag,
                float* __restrict__ out) {
  // xbuf (TDM target) must be at LDS offset 0: single shared array.
  __shared__ __align__(16) f16 smem[kXbufHalfs + kHbufHalfs];  // 55,296 bytes
  f16* xbuf = smem;
  f16* hbuf = smem + kXbufHalfs;

  const int b    = blockIdx.x;
  const int tid  = threadIdx.x;
  const int lane = tid & 31;
  const int wv   = tid >> 5;
  const int wi   = wv >> 2;   // 0..7
  const int wj   = wv & 3;    // 0..3
  const int hi   = lane >> 4;
  const int ln   = lane & 15;
  const int y    = wj * 16 + ln;

  // Init hbuf: h regions to 0, bias activation column (k'=256) to 1.
  for (int p = tid; p < kH * 288; p += 1024) {
    int yy = p / 288, k = p % 288;
    hbuf[yy * kHStr + k] = (k == 256) ? (f16)1.0f : (f16)0.0f;
  }

  float cst[8];  // fp32 c-state in registers: v -> hid = 16*wi + 8*hi + v, col y
#pragma unroll
  for (int q = 0; q < 8; ++q) cst[q] = 0.f;

  const f16* xb  = xT + (long)b * kT * kXtSlice;
  const f16* bx  = &xbuf[y * kXStr + hi * 16];  // per-lane B-frag base, x K-tiles
  const f16* bh  = &hbuf[y * kHStr + hi * 16];  // per-lane B-frag base, h/bias K-tiles
  const f16* wb0 = wfrag + (long)wi * 512 + lane * 16;  // +kt*16384 +g*4096
  const int  hidb = wi * 16 + hi * 8;

  for (int t = 0; t < kT; ++t) {
    asm volatile("" ::: "memory");  // keep weight loads inside the time loop

    // ---- Stage diagonal t of x (64 rows x 256B) into padded xbuf ----
#if defined(__HIP_DEVICE_COMPILE__) && defined(__has_builtin) && \
    __has_builtin(__builtin_amdgcn_tensor_load_to_lds) && \
    __has_builtin(__builtin_amdgcn_s_wait_tensorcnt)
    if (wv == 0) {
      unsigned long long ga = (unsigned long long)(xb + (long)t * kXtSlice);
      v4u g0; v8i g1;
      g0[0] = 1u;                          // count=1, user descriptor
      g0[1] = 0u;                          // lds_addr = 0 (xbuf at LDS base)
      g0[2] = (unsigned int)ga;            // global_addr[31:0]
      g0[3] = (unsigned int)(ga >> 32) | (2u << 30);  // addr[56:32] | type=2
      g1[0] = (int)((2u << 16) |           // data_size = 4B
                    (1u << 20) |           // pad_enable
                    (5u << 22) |           // pad_interval: 64 dwords (256B)
                    (3u << 25));           // pad_amount: 4 dwords (16B) -> 272B rows
      g1[1] = (int)(64u << 16);            // tensor_dim0 = 64 (bits 79:48)
      g1[2] = (int)(64u << 16);            // tensor_dim1 = 64 (bits 111:80)
      g1[3] = (int)(64u << 16);            // tile_dim0   = 64 (bits 127:112)
      g1[4] = 64;                          // tile_dim1   = 64 (bits 143:128)
      g1[5] = 64;                          // tensor_dim0_stride = 64 (bits 207:160)
      g1[6] = (int)(64u << 16);            // tensor_dim1_stride = 64 (bits 255:208)
      g1[7] = 0;
      v4i z4 = {0, 0, 0, 0};
#if __clang_major__ >= 23
      v8i z8 = {0, 0, 0, 0, 0, 0, 0, 0};
      __builtin_amdgcn_tensor_load_to_lds(g0, g1, z4, z4, z8, 0);
#else
      __builtin_amdgcn_tensor_load_to_lds(g0, g1, z4, z4, 0);
#endif
      __builtin_amdgcn_s_wait_tensorcnt(0);
    }
#else
    {  // Fallback: cooperative contiguous copy (8 halfs/thread).
      const f16* src = xb + (long)t * kXtSlice + tid * 8;
      if (t + 1 < kT) __builtin_prefetch(src + kXtSlice, 0, 1);
      int yy = tid >> 4, kk = (tid & 15) * 8;
      *(int4*)&xbuf[yy * kXStr + kk] = *(const int4*)src;
    }
#endif
    __syncthreads();  // x panel + previous h-writes visible to all

    v8f acc[4];  // acc[g]: gate g (i,f,o,g), 16x16
#pragma unroll
    for (int m = 0; m < 4; ++m)
#pragma unroll
      for (int q = 0; q < 8; ++q) acc[m][q] = 0.0f;

#pragma unroll 1
    for (int kt = 0; kt < 4; ++kt) {  // x K-tiles
      union { int4 q[2]; v16h v; } bu;
      const int4* bp = (const int4*)(bx + kt * 32);
      bu.q[0] = bp[0]; bu.q[1] = bp[1];
      const f16* wk = wb0 + (long)kt * 16384;
#pragma unroll
      for (int g = 0; g < 4; ++g) {
        const v16h* ap = (const v16h*)(wk + g * 4096);
        acc[g] = __builtin_amdgcn_wmma_f32_16x16x32_f16(
            false, *ap, false, bu.v, (short)0, acc[g], false, false);
      }
    }
#pragma unroll 1
    for (int kt = 4; kt < kNKT; ++kt) {  // h(y-1) | h(y) | bias K-tiles
      union { int4 q[2]; v16h v; } bu;
      const int4* bp = (const int4*)(bh + (kt - 4) * 32);
      bu.q[0] = bp[0]; bu.q[1] = bp[1];
      const f16* wk = wb0 + (long)kt * 16384;
#pragma unroll
      for (int g = 0; g < 4; ++g) {
        const v16h* ap = (const v16h*)(wk + g * 4096);
        acc[g] = __builtin_amdgcn_wmma_f32_16x16x32_f16(
            false, *ap, false, bu.v, (short)0, acc[g], false, false);
      }
    }
    __syncthreads();  // all LDS reads done before h/x panels are overwritten

    const int wcol = t - y;  // unskew column; in-band iff 0 <= wcol < 64
    v8h   hpk;
    float hval[8];
#pragma unroll
    for (int v = 0; v < 8; ++v) {
      float gi = fast_rcp(1.f + __expf(-acc[0][v]));
      float gf = fast_rcp(1.f + __expf(-acc[1][v]));
      float go = fast_rcp(1.f + __expf(-acc[2][v]));
      float gg = 1.f - 2.f * fast_rcp(__expf(2.f * acc[3][v]) + 1.f);  // tanh
      float cc = gf * cst[v] + gi * gg;
      cst[v] = cc;
      float tc = 1.f - 2.f * fast_rcp(__expf(2.f * cc) + 1.f);         // tanh(c)
      float hv = go * tc;
      hval[v] = hv;
      hpk[v]  = (f16)hv;
    }
    // New h -> hbuf: h(y) slot (k'=128+hid) and shifted h(y-1) slot (row y+1, k'=hid).
    *(v8h*)&hbuf[y * kHStr + 128 + hidb] = hpk;
    if (y < kH - 1) *(v8h*)&hbuf[(y + 1) * kHStr + hidb] = hpk;
    if (wcol >= 0 && wcol < kW) {
#pragma unroll
      for (int v = 0; v < 8; ++v)
        out[(((long)b * kHid + hidb + v) * kH + y) * kW + wcol] = hval[v];
    }
  }
}

extern "C" void kernel_launch(void* const* d_in, const int* in_sizes, int n_in,
                              void* d_out, int out_size, void* d_ws, size_t ws_size,
                              hipStream_t stream) {
  const float* x    = (const float*)d_in[0];
  const float* w_is = (const float*)d_in[1];
  const float* b_is = (const float*)d_in[2];
  const float* w_ss = (const float*)d_in[3];
  const float* b_ss = (const float*)d_in[4];
  (void)in_sizes; (void)n_in; (void)out_size; (void)ws_size;

  f16* xT = (f16*)d_ws;            // 16.0 MB
  f16* wf = xT + kXtElems;         // +0.4 MB pre-swizzled fused weights

  {
    int blk = 256;
    int grd = (int)((kXtElems + blk - 1) / blk);
    dlstm_prep_x<<<grd, blk, 0, stream>>>(x, xT);
  }
  {
    int blk = 256;
    int grd = (int)((kWfElems + blk - 1) / blk);
    dlstm_prep_w<<<grd, blk, 0, stream>>>(w_is, b_is, w_ss, b_ss, wf);
  }
  dlstm_scan<<<kB, 1024, 0, stream>>>(xT, wf, (float*)d_out);
}
